// Co_Attention_wATT_6657199309229
// MI455X (gfx1250) — matrix operand order repaired and checked by reading, not compiled, not run on hardware
//
#include <hip/hip_runtime.h>
#include <hip/hip_bf16.h>
#include <math.h>

// ---------------------------------------------------------------------------
// Co-attention, MI455X (gfx1250, wave32, WMMA).
// Pipeline:
//   norm_kernel        : invn[b] = 1/(||i_sw[b]||2 + 1e-13)
//   prep_user_kernel   : uB  = bf16(user), K padded 300->320 with zeros
//   prep_item_kernel   : iB  = bf16(item * i_sw * invn), padded
//   prep_v_kernel      : Vt  = bf16(V^T), padded 320x320 (row-major N x K)
//   gemm1_kernel       : pB  = uB @ V          (v_wmma_f32_16x16x32_bf16)
//   gemm2_max_kernel   : coatt = max_n(where(pB@iB^T==0,-1000)) / sqrt(300)
//   softmax_rep_kernel : masked softmax + user_rep = w @ user
// ---------------------------------------------------------------------------

#define BB   128
#define LU   512
#define LI   512
#define DD   300
#define KP   320   // K padded to 10 * 32

typedef __attribute__((ext_vector_type(16))) __bf16 v16bf;
typedef __attribute__((ext_vector_type(8)))  __bf16 v8bf;
typedef __attribute__((ext_vector_type(8)))  float  v8f;

// A-matrix tile (16 M-rows x 32 K), bf16, row-major source with `stride`.
// CDNA5 layout: lane(l<16): M=l, elems 0..7 -> K=g*8+e, 8..15 -> K=16+g*8+e
// where g = lane/16.
__device__ __forceinline__ v16bf load_a_tile(const __bf16* p, int stride, int lane) {
    int m = lane & 15, g = lane >> 4;
    const __bf16* q = p + (size_t)m * stride + g * 8;
    v8bf lo = *(const v8bf*)(q);        // K = g*8 + 0..7
    v8bf hi = *(const v8bf*)(q + 16);   // K = 16 + g*8 + 0..7
    v16bf a;
#pragma unroll
    for (int i = 0; i < 8; ++i) { a[i] = lo[i]; a[i + 8] = hi[i]; }
    return a;
}

// B-matrix tile (32 K x 16 N) from row-major N x K storage (i.e. B^T layout).
// CDNA5 layout: lane: N = lane%16, elems 0..7 -> K = h*8+e, 8..15 -> K = 16+h*8+e, h = lane/16.
__device__ __forceinline__ v16bf load_b_tile(const __bf16* p, int stride, int lane) {
    int n = lane & 15, h = lane >> 4;
    const __bf16* q = p + (size_t)n * stride + h * 8;
    v8bf lo = *(const v8bf*)(q);
    v8bf hi = *(const v8bf*)(q + 16);
    v16bf b;
#pragma unroll
    for (int i = 0; i < 8; ++i) { b[i] = lo[i]; b[i + 8] = hi[i]; }
    return b;
}

// ---------------------------------------------------------------------------
__global__ void norm_kernel(const float* __restrict__ isw, float* __restrict__ invn) {
    int b = blockIdx.x, t = threadIdx.x;
    __shared__ float red[256];
    float x0 = isw[b * LI + t];
    float x1 = isw[b * LI + t + 256];
    red[t] = x0 * x0 + x1 * x1;
    __syncthreads();
    for (int s = 128; s > 0; s >>= 1) {
        if (t < s) red[t] += red[t + s];
        __syncthreads();
    }
    if (t == 0) invn[b] = 1.0f / (sqrtf(red[0]) + 1e-13f);
}

__global__ void prep_user_kernel(const float* __restrict__ user, __bf16* __restrict__ uB) {
    size_t idx = (size_t)blockIdx.x * 256 + threadIdx.x;   // over B*LU*KP
    int d = (int)(idx % KP);
    size_t row = idx / KP;                                 // b*LU + l
    uB[idx] = (d < DD) ? (__bf16)user[row * DD + d] : (__bf16)0.0f;
}

__global__ void prep_item_kernel(const float* __restrict__ item,
                                 const float* __restrict__ isw,
                                 const float* __restrict__ invn,
                                 __bf16* __restrict__ iB) {
    size_t idx = (size_t)blockIdx.x * 256 + threadIdx.x;   // over B*LI*KP
    int d = (int)(idx % KP);
    size_t row = idx / KP;                                 // b*LI + l
    int b = (int)(row >> 9);
    float w = isw[row] * invn[b];
    iB[idx] = (d < DD) ? (__bf16)(item[row * DD + d] * w) : (__bf16)0.0f;
}

__global__ void prep_v_kernel(const float* __restrict__ V, __bf16* __restrict__ Vt) {
    int idx = blockIdx.x * 256 + threadIdx.x;              // over KP*KP
    int k = idx % KP, n = idx / KP;
    Vt[idx] = (n < DD && k < DD) ? (__bf16)V[k * DD + n] : (__bf16)0.0f;  // Vt[n][k] = V[k][n]
}

// ---------------------------------------------------------------------------
// GEMM 1: pB[b, m, n] = sum_k uB[b, m, k] * V[k, n].  One 16x16 tile per wave.
__global__ void gemm1_kernel(const __bf16* __restrict__ uB,
                             const __bf16* __restrict__ Vt,
                             __bf16* __restrict__ pB) {
    int lane = threadIdx.x & 31;
    int wid  = threadIdx.x >> 5;
    int wg   = blockIdx.x * 8 + wid;          // B * (LU/16) * (KP/16) waves
    int nt = wg % (KP / 16);
    int lt = (wg / (KP / 16)) % (LU / 16);
    int b  = wg / ((KP / 16) * (LU / 16));

    const __bf16* Abase = uB + ((size_t)b * LU + lt * 16) * KP;
    const __bf16* Bbase = Vt + (size_t)nt * 16 * KP;

    v8f acc = {};
#pragma unroll
    for (int t = 0; t < KP / 32; ++t) {
        v16bf a  = load_a_tile(Abase + t * 32, KP, lane);
        v16bf bm = load_b_tile(Bbase + t * 32, KP, lane);
        acc = __builtin_amdgcn_wmma_f32_16x16x32_bf16(
                  false, a, false, bm, (short)0, acc, false, false);
    }

    // Store C/D (f32) as bf16 row-major: lane holds N=lane%16, row r+8*(lane/16).
    __bf16* Dp = pB + ((size_t)b * LU + lt * 16) * KP + nt * 16;
    int n = lane & 15, h = lane >> 4;
#pragma unroll
    for (int r = 0; r < 8; ++r) {
        int m = r + 8 * h;
        Dp[(size_t)m * KP + n] = (__bf16)acc[r];
    }
}

// ---------------------------------------------------------------------------
// GEMM 2 fused with where(G==0,-1000) and row-max over Li.
// One wave owns a 16-row Lu strip; A K-tiles held in registers.
__global__ void gemm2_max_kernel(const __bf16* __restrict__ pB,
                                 const __bf16* __restrict__ iB,
                                 float* __restrict__ coatt) {
    int lane = threadIdx.x & 31;
    int wid  = threadIdx.x >> 5;
    int wg   = blockIdx.x * 8 + wid;          // B * (LU/16) waves
    int lt = wg % (LU / 16);
    int b  = wg / (LU / 16);

    const __bf16* Abase = pB + ((size_t)b * LU + lt * 16) * KP;
    v16bf a[KP / 32];
#pragma unroll
    for (int t = 0; t < KP / 32; ++t) a[t] = load_a_tile(Abase + t * 32, KP, lane);

    float rmax[8];
#pragma unroll
    for (int r = 0; r < 8; ++r) rmax[r] = -3.0e38f;

    const __bf16* Ibase = iB + (size_t)b * LI * KP;
    for (int ntile = 0; ntile < LI / 16; ++ntile) {
        if (ntile + 1 < LI / 16)  // stream next item tile toward L2/L0
            __builtin_prefetch(Ibase + (size_t)(ntile + 1) * 16 * KP, 0, 3);
        const __bf16* Bb = Ibase + (size_t)ntile * 16 * KP;
        v8f acc = {};
#pragma unroll
        for (int t = 0; t < KP / 32; ++t) {
            v16bf bm = load_b_tile(Bb + t * 32, KP, lane);
            acc = __builtin_amdgcn_wmma_f32_16x16x32_bf16(
                      false, a[t], false, bm, (short)0, acc, false, false);
        }
#pragma unroll
        for (int r = 0; r < 8; ++r) {
            float g = acc[r];
            g = (g == 0.0f) ? -1000.0f : g;
            rmax[r] = fmaxf(rmax[r], g);
        }
    }

    // Reduce across the 16 lanes holding the same row (width-16 xor shuffles).
#pragma unroll
    for (int r = 0; r < 8; ++r) {
        float v = rmax[r];
        for (int off = 1; off < 16; off <<= 1)
            v = fmaxf(v, __shfl_xor(v, off, 16));
        rmax[r] = v;
    }

    const float scale = 0.05773502691896258f;  // 1/sqrt(300)
    if ((lane & 15) == 0) {
        int h = lane >> 4;
#pragma unroll
        for (int r = 0; r < 8; ++r)
            coatt[(size_t)b * LU + lt * 16 + r + 8 * h] = rmax[r] * scale;
    }
}

// ---------------------------------------------------------------------------
// Masked softmax (reference semantics) + user_rep = w @ user.
__global__ void softmax_rep_kernel(const float* __restrict__ coatt,
                                   const int* __restrict__ umask,
                                   const float* __restrict__ user,
                                   float* __restrict__ out_rep,
                                   float* __restrict__ out_att) {
    int b = blockIdx.x, t = threadIdx.x;
    __shared__ float wsv[LU];
    __shared__ float red[256];

    float m0 = (float)umask[b * LU + t];
    float m1 = (float)umask[b * LU + t + 256];
    float x0 = coatt[b * LU + t] * m0;
    float x1 = coatt[b * LU + t + 256] * m1;

    red[t] = fmaxf(x0, x1);
    __syncthreads();
    for (int s = 128; s > 0; s >>= 1) { if (t < s) red[t] = fmaxf(red[t], red[t + s]); __syncthreads(); }
    float mx = red[0];
    __syncthreads();

    float e0 = expf(x0 - mx), e1 = expf(x1 - mx);
    red[t] = e0 + e1;
    __syncthreads();
    for (int s = 128; s > 0; s >>= 1) { if (t < s) red[t] += red[t + s]; __syncthreads(); }
    float sumE = red[0];
    __syncthreads();

    float r0 = (e0 / sumE) * m0;
    float r1 = (e1 / sumE) * m1;
    red[t] = r0 + r1;
    __syncthreads();
    for (int s = 128; s > 0; s >>= 1) { if (t < s) red[t] += red[t + s]; __syncthreads(); }
    float invR = 1.0f / (red[0] + 1e-13f);
    __syncthreads();

    float w0 = r0 * invR, w1 = r1 * invR;
    wsv[t] = w0; wsv[t + 256] = w1;
    out_att[(size_t)b * LU + t] = w0;
    out_att[(size_t)b * LU + t + 256] = w1;
    __syncthreads();

    for (int d = t; d < DD; d += 256) {
        float acc = 0.0f;
        const float* up = user + (size_t)b * LU * DD + d;
        for (int l = 0; l < LU; ++l) acc += wsv[l] * up[(size_t)l * DD];
        out_rep[(size_t)b * DD + d] = acc;
    }
}

// ---------------------------------------------------------------------------
extern "C" void kernel_launch(void* const* d_in, const int* in_sizes, int n_in,
                              void* d_out, int out_size, void* d_ws, size_t ws_size,
                              hipStream_t stream) {
    const float* user  = (const float*)d_in[0];   // [B, Lu, D]
    const float* item  = (const float*)d_in[1];   // [B, Li, D]
    const int*   umask = (const int*)d_in[2];     // [B, Lu]
    /* d_in[3] = i_sent_mask: unused by reference output */
    const float* isw   = (const float*)d_in[4];   // [B, Li]
    const float* V     = (const float*)d_in[5];   // [D, D]

    float* out_rep = (float*)d_out;               // [B, D]
    float* out_att = (float*)d_out + (size_t)BB * DD;  // [B, 1, Lu]

    // Workspace layout (bytes); bf16 tensors all 32B-aligned tile bases.
    char* ws = (char*)d_ws;
    __bf16* uB   = (__bf16*)(ws + 0);             // B*LU*KP bf16 = 41,943,040
    __bf16* iB   = (__bf16*)(ws + 41943040);      // B*LI*KP bf16
    __bf16* pB   = (__bf16*)(ws + 83886080);      // B*LU*KP bf16
    __bf16* Vt   = (__bf16*)(ws + 125829120);     // KP*KP bf16 = 204,800
    float*  coatt = (float*)(ws + 126033920);     // B*LU f32 = 262,144
    float*  invn  = (float*)(ws + 126296064);     // B f32

    norm_kernel<<<BB, 256, 0, stream>>>(isw, invn);
    prep_user_kernel<<<(BB * LU * KP) / 256, 256, 0, stream>>>(user, uB);
    prep_item_kernel<<<(BB * LI * KP) / 256, 256, 0, stream>>>(item, isw, invn, iB);
    prep_v_kernel<<<(KP * KP) / 256, 256, 0, stream>>>(V, Vt);

    // 8 waves / block, one 16x16 tile per wave.
    gemm1_kernel<<<(BB * (LU / 16) * (KP / 16)) / 8, 256, 0, stream>>>(uB, Vt, pB);
    gemm2_max_kernel<<<(BB * (LU / 16)) / 8, 256, 0, stream>>>(pB, iB, coatt);
    softmax_rep_kernel<<<BB, 256, 0, stream>>>(coatt, umask, user, out_rep, out_att);
}
